// MultiQueryAttention_39333310496900
// MI455X (gfx1250) — matrix operand order repaired and checked by reading, compiled, not run on hardware
//
#include <hip/hip_runtime.h>
#include <math.h>

#define D_MODEL   2048
#define HEAD_DIM  128
#define NUM_HEADS 16

typedef __attribute__((ext_vector_type(16))) __bf16         v16bf;
typedef __attribute__((ext_vector_type(8)))  float          v8f;
typedef __attribute__((ext_vector_type(8)))  unsigned short v8u16;
typedef __attribute__((ext_vector_type(4)))  unsigned int   v4u32;
typedef __attribute__((ext_vector_type(4)))  unsigned int   u32x4;
typedef __attribute__((ext_vector_type(4)))  int            i32x4;
typedef __attribute__((ext_vector_type(8)))  int            i32x8;
typedef unsigned short u16;
typedef unsigned int   u32;
typedef unsigned long long u64;

#if defined(__AMDGCN__) && __has_builtin(__builtin_amdgcn_tensor_load_to_lds)
#define USE_TDM 1
#else
#define USE_TDM 0
#endif

__device__ __forceinline__ u16 f2bf(float f) {
  u32 u = __builtin_bit_cast(u32, f);
  u += 0x7FFFu + ((u >> 16) & 1u);   // round-to-nearest-even
  return (u16)(u >> 16);
}

__device__ __forceinline__ v16bf make_frag(v8u16 lo, v8u16 hi) {
  union { v8u16 u[2]; v16bf v; } x;
  x.u[0] = lo; x.u[1] = hi;
  return x.v;
}

__device__ __forceinline__ void wait_tensorcnt0() {
#if defined(__AMDGCN__) && __has_builtin(__builtin_amdgcn_s_wait_tensorcnt)
  __builtin_amdgcn_s_wait_tensorcnt((short)0);
#elif defined(__AMDGCN__)
  asm volatile("s_wait_tensorcnt 0x0" ::: "memory");
#endif
}

#if USE_TDM
// Issue a 2D TDM tile load (global -> LDS) per CDNA5 D# layout (ISA 08 §8).
// Sizes/strides in elements of data_size=2B. Optional LDS row padding.
__device__ __forceinline__ void tdm_load_2d(
    u32 lds_addr, const u16* gptr,
    u32 tensor_dim0, u32 tensor_dim1,
    u32 tile_dim0, u32 tile_dim1,
    u64 stride0, u32 pad_interval_code, u32 pad_amount_code, int pad_en) {
  u64 ga = (u64)(uintptr_t)gptr;
  u32x4 g0;
  g0[0] = 1u;                                   // count=1, user descriptor
  g0[1] = lds_addr;                             // lds byte address
  g0[2] = (u32)ga;                              // global_addr[31:0]
  g0[3] = (u32)((ga >> 32) & 0x01FFFFFFu) | (2u << 30);  // addr[56:32]|type=2
  i32x8 g1;
  g1[0] = (int)((1u << 16) |                    // data_size = 2 bytes
                ((u32)(pad_en ? 1u : 0u) << 20) |
                (pad_interval_code << 22) | (pad_amount_code << 25));
  g1[1] = (int)((tensor_dim0 & 0xFFFFu) << 16);                 // dim0 lo
  g1[2] = (int)((tensor_dim0 >> 16) | ((tensor_dim1 & 0xFFFFu) << 16));
  g1[3] = (int)((tensor_dim1 >> 16) | (tile_dim0 << 16));
  g1[4] = (int)(tile_dim1 & 0xFFFFu);                           // tile_dim1
  g1[5] = (int)(u32)stride0;                                    // stride lo
  g1[6] = (int)(u32)((stride0 >> 32) & 0xFFFFu);                // stride hi
  g1[7] = 0;
  i32x4 z4 = {0, 0, 0, 0};
#if __clang_major__ >= 23
  i32x8 z8 = {0, 0, 0, 0, 0, 0, 0, 0};
  __builtin_amdgcn_tensor_load_to_lds(g0, g1, z4, z4, z8, 0);
#else
  __builtin_amdgcn_tensor_load_to_lds(g0, g1, z4, z4, 0);
#endif
}
#endif

// ---------------------------------------------------------------- f32 -> bf16
__global__ __launch_bounds__(256) void cvt_bf16_kernel(
    const float* __restrict__ in, u16* __restrict__ out, long n) {
  long i = ((long)blockIdx.x * blockDim.x + threadIdx.x) * 4;
  if (i + 3 < n) {
    float4 f = *(const float4*)(in + i);
    union { u16 s[4]; unsigned long long q; } o;
    o.s[0] = f2bf(f.x); o.s[1] = f2bf(f.y);
    o.s[2] = f2bf(f.z); o.s[3] = f2bf(f.w);
    *(unsigned long long*)(out + i) = o.q;
  }
}

// ------------------------------------------------------- bf16 WMMA GEMM + bias
// C[M,N] = A[M,K] (bf16 row-major) * Bw[K,N] (bf16 row-major) + bias
#define BM 128
#define BN 128
#define BK 64
#define AS_STRIDE 72    // 64 + 8 pad  (pad: 16B after each 128B row)
#define BS_STRIDE 136   // 128 + 8 pad (pad: 16B after each 256B row)

template <bool OUT_BF16>
__global__ __launch_bounds__(256, 1) void gemm_bf16_wmma(
    const u16* __restrict__ A, const u16* __restrict__ Bw,
    const float* __restrict__ bias,
    u16* __restrict__ Cb, float* __restrict__ Cf,
    int M, int N, int K) {
  __shared__ u16 As[2][BM * AS_STRIDE];
  __shared__ u16 Bs[2][BK * BS_STRIDE];

  const int tid  = threadIdx.x;
  const int wave = tid >> 5;
  const int lane = tid & 31;
  const int h    = lane >> 4;
  const int ml   = lane & 15;
  const int kl   = ml + 16 * h;          // unused; B fragment holds k = 8h+e and 16+8h+e
  const int waveM = (wave >> 1) * 32;
  const int waveN = (wave & 1) * 64;
  const long tileM = (long)blockIdx.y * BM;
  const long tileN = (long)blockIdx.x * BN;
  const int nIter = K / BK;

  v8f acc[2][4];
  for (int i = 0; i < 2; ++i)
    for (int j = 0; j < 4; ++j)
      for (int e = 0; e < 8; ++e) acc[i][j][e] = 0.f;

#if USE_TDM
  if (wave == 0) {
    tdm_load_2d((u32)(uintptr_t)&As[0][0], &A[tileM * (long)K], K, M, BK, BM,
                (u64)K, 4, 3, 1);
    tdm_load_2d((u32)(uintptr_t)&Bs[0][0], &Bw[tileN], N, K, BN, BK,
                (u64)N, 5, 3, 1);
    wait_tensorcnt0();
  }
  __syncthreads();
#endif

  for (int it = 0; it < nIter; ++it) {
    const int cur = it & 1;
#if USE_TDM
    if (it + 1 < nIter && wave == 0) {
      const long k0 = (long)(it + 1) * BK;
      tdm_load_2d((u32)(uintptr_t)&As[cur ^ 1][0], &A[tileM * (long)K + k0],
                  K, M, BK, BM, (u64)K, 4, 3, 1);
      tdm_load_2d((u32)(uintptr_t)&Bs[cur ^ 1][0], &Bw[k0 * N + tileN],
                  N, K, BN, BK, (u64)N, 5, 3, 1);
    }
#else
    {
      const long k0 = (long)it * BK;
      __syncthreads();
      #pragma unroll
      for (int c = 0; c < 4; ++c) {               // A tile: 1024 16B chunks
        int idx = tid + c * 256;
        int row = idx >> 3;
        int col = (idx & 7) * 8;
        *(v4u32*)&As[cur][row * AS_STRIDE + col] =
            *(const v4u32*)&A[(tileM + row) * (long)K + k0 + col];
      }
      #pragma unroll
      for (int c = 0; c < 4; ++c) {               // B tile: 1024 16B chunks
        int idx = tid + c * 256;
        int row = idx >> 4;
        int col = (idx & 15) * 8;
        *(v4u32*)&Bs[cur][row * BS_STRIDE + col] =
            *(const v4u32*)&Bw[(k0 + row) * (long)N + tileN + col];
      }
      __syncthreads();
    }
#endif

    const u16* As_c = &As[cur][0];
    const u16* Bs_c = &Bs[cur][0];
    v16bf af[2][2];
    #pragma unroll
    for (int i = 0; i < 2; ++i)
      #pragma unroll
      for (int c = 0; c < 2; ++c) {
        const u16* p = &As_c[(waveM + i * 16 + ml) * AS_STRIDE + c * 32 + h * 8];
        af[i][c] = make_frag(*(const v8u16*)p, *(const v8u16*)(p + 16));
      }
    #pragma unroll
    for (int c = 0; c < 2; ++c)
      #pragma unroll
      for (int j = 0; j < 4; ++j) {
        const u16* p = &Bs_c[(c * 32 + h * 8) * BS_STRIDE + waveN + j * 16 + ml];
        v8u16 blo, bhi;
        #pragma unroll
        for (int e = 0; e < 8; ++e) { blo[e] = p[e * BS_STRIDE]; bhi[e] = p[(16 + e) * BS_STRIDE]; }
        v16bf bf = make_frag(blo, bhi);
        #pragma unroll
        for (int i = 0; i < 2; ++i)
          acc[i][j] = __builtin_amdgcn_wmma_f32_16x16x32_bf16(
              false, af[i][c], false, bf, (short)0, acc[i][j], false, false);
      }

#if USE_TDM
    if (wave == 0) wait_tensorcnt0();
    __syncthreads();
#endif
  }

  #pragma unroll
  for (int j = 0; j < 4; ++j) {
    const int col = (int)tileN + waveN + j * 16 + ml;
    const float bv = bias[col];
    #pragma unroll
    for (int i = 0; i < 2; ++i)
      #pragma unroll
      for (int r = 0; r < 8; ++r) {
        long row = tileM + waveM + i * 16 + h * 8 + r;
        float v = acc[i][j][r] + bv;
        if (OUT_BF16) Cb[row * N + col] = f2bf(v);
        else          Cf[row * N + col] = v;
      }
  }
}

// ------------------------------------------------- flash MQA attention (bf16)
#define KT 128
#define QT 128
#define KSTR 136

__global__ __launch_bounds__(256, 1) void mqa_flash_wmma(
    const u16* __restrict__ Q,   // [B,S,D_MODEL]
    const u16* __restrict__ Kp,  // [B,S,HEAD_DIM]
    const u16* __restrict__ Vp,  // [B,S,HEAD_DIM]
    u16* __restrict__ O,         // [B,S,D_MODEL]
    int S) {
  __shared__ u16 Kt[HEAD_DIM * KSTR];   // K transposed: [d][key]
  __shared__ u16 Vs[KT * KSTR];         // V direct:     [key][d]
  __shared__ u16 Ps[8 * 16 * KSTR];     // per-wave P:   [qrow][key]

  const int tid  = threadIdx.x;
  const int wave = tid >> 5;
  const int lane = tid & 31;
  const int h    = lane >> 4;
  const int ml   = lane & 15;
  const int kl   = ml + 16 * h;
  const int b    = blockIdx.z;
  const int head = blockIdx.y;
  const int qt   = blockIdx.x;
  const long qbase = (long)b * S + (long)qt * QT;
  const float scale = 0.08838834764831845f;   // 1/sqrt(128)

  // Q fragments held in registers for the whole kernel (A-operand layout)
  v16bf qf[4];
  {
    const u16* qrowp = Q + (qbase + wave * 16 + ml) * D_MODEL + head * HEAD_DIM;
    #pragma unroll
    for (int c = 0; c < 4; ++c) {
      const u16* p = qrowp + c * 32 + h * 8;
      qf[c] = make_frag(*(const v8u16*)p, *(const v8u16*)(p + 16));
    }
  }

  v8f oacc[8];
  for (int j = 0; j < 8; ++j)
    for (int e = 0; e < 8; ++e) oacc[j][e] = 0.f;
  float mrow[8], lrow[8];
  for (int r = 0; r < 8; ++r) { mrow[r] = -1e30f; lrow[r] = 0.f; }

  u16* Pw = &Ps[wave * 16 * KSTR];

  for (int k0 = 0; k0 < S; k0 += KT) {
    const u16* Kg = Kp + ((long)b * S + k0) * HEAD_DIM;
    const u16* Vg = Vp + ((long)b * S + k0) * HEAD_DIM;

#if USE_TDM
    if (wave == 0)      // V tile via Tensor Data Mover (no transpose needed)
      tdm_load_2d((u32)(uintptr_t)&Vs[0], Vg, HEAD_DIM, KT, HEAD_DIM, KT,
                  (u64)HEAD_DIM, 5, 3, 1);
#else
    for (int it = 0; it < 8; ++it) {
      int idx = tid + it * 256;
      int key = idx >> 4;
      int dd  = (idx & 15) * 8;
      *(v4u32*)&Vs[key * KSTR + dd] =
          *(const v4u32*)&Vg[(long)key * HEAD_DIM + dd];
    }
#endif
    // K transposed into LDS: vector global reads, scalar scattered DS writes
    for (int it = 0; it < 8; ++it) {
      int idx = tid + it * 256;               // 2048 chunks of 8 elements
      int key = idx >> 4;
      int dd0 = (idx & 15) * 8;
      v8u16 kv = *(const v8u16*)&Kg[(long)key * HEAD_DIM + dd0];
      #pragma unroll
      for (int e = 0; e < 8; ++e)
        Kt[(dd0 + e) * KSTR + key] = kv[e];
    }
#if USE_TDM
    if (wave == 0) wait_tensorcnt0();
#endif
    __syncthreads();

    // scores S = Q * K^T  (8 key tiles of 16; K-dim = HEAD_DIM in 4 chunks)
    v8f sacc[8];
    for (int j = 0; j < 8; ++j)
      for (int e = 0; e < 8; ++e) sacc[j][e] = 0.f;
    #pragma unroll
    for (int j = 0; j < 8; ++j)
      #pragma unroll
      for (int c = 0; c < 4; ++c) {
        const u16* p = &Kt[(c * 32 + h * 8) * KSTR + j * 16 + ml];
        v8u16 blo, bhi;
        #pragma unroll
        for (int e = 0; e < 8; ++e) { blo[e] = p[e * KSTR]; bhi[e] = p[(16 + e) * KSTR]; }
        v16bf bf = make_frag(blo, bhi);
        sacc[j] = __builtin_amdgcn_wmma_f32_16x16x32_bf16(
            false, qf[c], false, bf, (short)0, sacc[j], false, false);
      }

    // online softmax; row = wave*16 + 8h + r, cols across 16-lane groups
    #pragma unroll
    for (int r = 0; r < 8; ++r) {
      float mx = -1e30f;
      #pragma unroll
      for (int j = 0; j < 8; ++j) {
        float s = sacc[j][r] * scale;
        sacc[j][r] = s;
        mx = fmaxf(mx, s);
      }
      #pragma unroll
      for (int off = 1; off < 16; off <<= 1)
        mx = fmaxf(mx, __shfl_xor(mx, off, 32));
      float mnew = fmaxf(mrow[r], mx);
      float corr = __expf(mrow[r] - mnew);
      mrow[r] = mnew;
      float rs = 0.f;
      #pragma unroll
      for (int j = 0; j < 8; ++j) {
        float p = __expf(sacc[j][r] - mnew);
        sacc[j][r] = p;
        rs += p;
      }
      #pragma unroll
      for (int off = 1; off < 16; off <<= 1)
        rs += __shfl_xor(rs, off, 32);
      lrow[r] = lrow[r] * corr + rs;
      #pragma unroll
      for (int j = 0; j < 8; ++j) oacc[j][r] *= corr;
    }

    // P (C layout) -> per-wave LDS patch -> A-fragment layout
    #pragma unroll
    for (int j = 0; j < 8; ++j)
      #pragma unroll
      for (int r = 0; r < 8; ++r)
        Pw[(h * 8 + r) * KSTR + j * 16 + ml] = f2bf(sacc[j][r]);

    // O += P * V
    #pragma unroll
    for (int c = 0; c < 4; ++c) {
      const u16* pp = &Pw[ml * KSTR + c * 32 + h * 8];
      v16bf pf = make_frag(*(const v8u16*)pp, *(const v8u16*)(pp + 16));
      #pragma unroll
      for (int j = 0; j < 8; ++j) {
        const u16* vp = &Vs[(c * 32 + h * 8) * KSTR + j * 16 + ml];
        v8u16 vlo, vhi;
        #pragma unroll
        for (int e = 0; e < 8; ++e) { vlo[e] = vp[e * KSTR]; vhi[e] = vp[(16 + e) * KSTR]; }
        v16bf vf = make_frag(vlo, vhi);
        oacc[j] = __builtin_amdgcn_wmma_f32_16x16x32_bf16(
            false, pf, false, vf, (short)0, oacc[j], false, false);
      }
    }
    __syncthreads();
  }

  // normalize + store bf16 attn output at [b, s, head*128 + d]
  #pragma unroll
  for (int r = 0; r < 8; ++r) {
    float inv = 1.f / lrow[r];
    long row = qbase + wave * 16 + h * 8 + r;
    u16* orow = O + row * D_MODEL + head * HEAD_DIM;
    #pragma unroll
    for (int j = 0; j < 8; ++j)
      orow[j * 16 + ml] = f2bf(oacc[j][r] * inv);
  }
}

// ---------------------------------------------------------------------- host
extern "C" void kernel_launch(void* const* d_in, const int* in_sizes, int n_in,
                              void* d_out, int out_size, void* d_ws, size_t ws_size,
                              hipStream_t stream) {
  const float* q  = (const float*)d_in[0];
  const float* k  = (const float*)d_in[1];
  const float* v  = (const float*)d_in[2];
  const float* Wq = (const float*)d_in[3];
  const float* bq = (const float*)d_in[4];
  const float* Wk = (const float*)d_in[5];
  const float* bk = (const float*)d_in[6];
  const float* Wv = (const float*)d_in[7];
  const float* bv = (const float*)d_in[8];
  const float* Wo = (const float*)d_in[9];
  const float* bo = (const float*)d_in[10];

  const int  Bb = 2, S = 2048, D = D_MODEL, dh = HEAD_DIM;
  const long M  = (long)Bb * S;   // 4096

  char* w = (char*)d_ws;
  auto alloc = [&](size_t bytes) -> u16* {
    char* p = w;
    w += (bytes + 255) & ~(size_t)255;
    return (u16*)p;
  };
  u16* qx  = alloc((size_t)M * D * 2);
  u16* kx  = alloc((size_t)M * D * 2);
  u16* vx  = alloc((size_t)M * D * 2);
  u16* wqb = alloc((size_t)D * D * 2);
  u16* wkb = alloc((size_t)D * dh * 2);
  u16* wvb = alloc((size_t)D * dh * 2);
  u16* wob = alloc((size_t)D * D * 2);
  u16* Qp  = alloc((size_t)M * D * 2);
  u16* Kpp = alloc((size_t)M * dh * 2);
  u16* Vpp = alloc((size_t)M * dh * 2);
  u16* Ab  = alloc((size_t)M * D * 2);

  auto cvt = [&](const float* in, u16* out, long n) {
    cvt_bf16_kernel<<<dim3((unsigned)((n / 4 + 255) / 256)), dim3(256), 0, stream>>>(in, out, n);
  };
  cvt(q, qx, M * D);
  cvt(k, kx, M * D);
  cvt(v, vx, M * D);
  cvt(Wq, wqb, (long)D * D);
  cvt(Wk, wkb, (long)D * dh);
  cvt(Wv, wvb, (long)D * dh);
  cvt(Wo, wob, (long)D * D);

  dim3 blk(256);
  dim3 gQ(D / BN, (unsigned)(M / BM));    // (16, 32)
  gemm_bf16_wmma<true><<<gQ, blk, 0, stream>>>(qx, wqb, bq, Qp, nullptr, (int)M, D, D);
  dim3 gKV(dh / BN, (unsigned)(M / BM));  // (1, 32)
  gemm_bf16_wmma<true><<<gKV, blk, 0, stream>>>(kx, wkb, bk, Kpp, nullptr, (int)M, dh, D);
  gemm_bf16_wmma<true><<<gKV, blk, 0, stream>>>(vx, wvb, bv, Vpp, nullptr, (int)M, dh, D);

  dim3 gA(S / QT, NUM_HEADS, Bb);         // (16, 16, 2)
  mqa_flash_wmma<<<gA, blk, 0, stream>>>(Qp, Kpp, Vpp, Ab, S);

  gemm_bf16_wmma<false><<<gQ, blk, 0, stream>>>(Ab, wob, bo, nullptr, (float*)d_out, (int)M, D, D);
}
